// RNN_8151847928457
// MI455X (gfx1250) — matrix operand order repaired and checked by reading, not compiled, not run on hardware
//
#include <hip/hip_runtime.h>
#include <hip/hip_bf16.h>
#include <math.h>

// Problem sizes (fixed by the reference)
#define T_STEPS 512
#define BATCH   128
#define IN_DIM  512
#define HID     1024

typedef __attribute__((ext_vector_type(16))) __bf16 v16bf;
typedef __attribute__((ext_vector_type(8)))  float  v8f;

union Frag16 {               // 16 bf16 = 32 bytes, matches WMMA A/B operand
    uint4 q[2];
    v16bf v;
};

// A fragment (16x32 bf16) built from f32 row data: 16 contiguous elems
// at k = half*16+{0..15}, matching the contiguous B fragment.
__device__ __forceinline__ v16bf load_a_from_f32(const float* __restrict__ Arow,
                                                 int k0, int half)
{
    const float4 f0 = *(const float4*)(Arow + k0 + half * 16);
    const float4 f1 = *(const float4*)(Arow + k0 + half * 16 + 4);
    const float4 f2 = *(const float4*)(Arow + k0 + half * 16 + 8);
    const float4 f3 = *(const float4*)(Arow + k0 + half * 16 + 12);
    v16bf a;
    a[0]=(__bf16)f0.x;  a[1]=(__bf16)f0.y;  a[2]=(__bf16)f0.z;  a[3]=(__bf16)f0.w;
    a[4]=(__bf16)f1.x;  a[5]=(__bf16)f1.y;  a[6]=(__bf16)f1.z;  a[7]=(__bf16)f1.w;
    a[8]=(__bf16)f2.x;  a[9]=(__bf16)f2.y;  a[10]=(__bf16)f2.z; a[11]=(__bf16)f2.w;
    a[12]=(__bf16)f3.x; a[13]=(__bf16)f3.y; a[14]=(__bf16)f3.z; a[15]=(__bf16)f3.w;
    return a;
}

// B fragment: 16 contiguous bf16 (k = half*16+{0..15}) for column N = lane&15.
__device__ __forceinline__ Frag16 load_b_global(const unsigned short* __restrict__ p)
{
    Frag16 b;
    b.q[0] = *(const uint4*)(p);
    b.q[1] = *(const uint4*)(p + 8);
    return b;
}

// ---------------------------------------------------------------------------
// Prep: bias = bx+bh; h0 -> bf16; WxT[n][i] = bf16(Wx[i][n]); WhT[n][k] = bf16(Wh[k][n])
// ---------------------------------------------------------------------------
__global__ __launch_bounds__(256) void rnn_prep_kernel(
    const float* __restrict__ Wx, const float* __restrict__ bx,
    const float* __restrict__ Wh, const float* __restrict__ bh,
    const float* __restrict__ h0,
    float* __restrict__ bias, unsigned short* __restrict__ hb0,
    unsigned short* __restrict__ WxT, unsigned short* __restrict__ WhT)
{
    const int i = blockIdx.x * blockDim.x + threadIdx.x;   // grid covers 1M threads

    if (i < HID * HID) {            // WhT (transpose + cvt)
        const int n = i >> 10;
        const int k = i & 1023;
        reinterpret_cast<__hip_bfloat16*>(WhT)[i] = __float2bfloat16(Wh[(size_t)k * HID + n]);
    }
    if (i < HID * IN_DIM) {         // WxT (transpose + cvt)
        const int n  = i >> 9;
        const int kk = i & 511;
        reinterpret_cast<__hip_bfloat16*>(WxT)[i] = __float2bfloat16(Wx[(size_t)kk * HID + n]);
    }
    if (i < HID) bias[i] = bx[i] + bh[i];
    if (i < BATCH * HID)
        reinterpret_cast<__hip_bfloat16*>(hb0)[i] = __float2bfloat16(h0[i]);
}

// ---------------------------------------------------------------------------
// Phase 1: XP[m][n] = sum_i x[m][i] * WxT[n][i] + bias[n]    (m = t*B+b)
// One wave computes a 16x64 strip (A reused over 4 N-tiles). bf16 WMMA, f32 acc.
// Software-pipelined: chunk c+1 fragments are fetched before chunk c's WMMAs.
// ---------------------------------------------------------------------------
__global__ __launch_bounds__(256) void rnn_gemm1_kernel(
    const float* __restrict__ x, const unsigned short* __restrict__ WxT,
    const float* __restrict__ bias, float* __restrict__ XP)
{
    const int lane = threadIdx.x & 31;
    const int wv   = threadIdx.x >> 5;
    const int w    = blockIdx.x * 8 + wv;     // 0..65535
    const int mstrip = w >> 4;                // 0..4095  (16 rows each)
    const int nstrip = w & 15;                // 0..15    (64 cols each)
    const int half = lane >> 4;
    const int nl   = lane & 15;

    const float* Arow = x + (size_t)(mstrip * 16 + nl) * IN_DIM;

    const unsigned short* Bb0 = WxT + (size_t)((nstrip * 4 + 0) * 16 + nl) * IN_DIM + half * 16;
    const unsigned short* Bb1 = WxT + (size_t)((nstrip * 4 + 1) * 16 + nl) * IN_DIM + half * 16;
    const unsigned short* Bb2 = WxT + (size_t)((nstrip * 4 + 2) * 16 + nl) * IN_DIM + half * 16;
    const unsigned short* Bb3 = WxT + (size_t)((nstrip * 4 + 3) * 16 + nl) * IN_DIM + half * 16;

    v8f acc[4] = {{}, {}, {}, {}};

    // prologue: chunk 0 fragments
    v16bf  aC = load_a_from_f32(Arow, 0, half);
    Frag16 b0C = load_b_global(Bb0);
    Frag16 b1C = load_b_global(Bb1);
    Frag16 b2C = load_b_global(Bb2);
    Frag16 b3C = load_b_global(Bb3);

#pragma unroll
    for (int c = 0; c < 16; ++c) {            // K chunks of 32
        v16bf  aN  = {};
        Frag16 b0N = {}, b1N = {}, b2N = {}, b3N = {};
        if (c < 15) {                          // fetch next chunk first (stays in flight)
            const int k1 = (c + 1) * 32;
            aN  = load_a_from_f32(Arow, k1, half);
            b0N = load_b_global(Bb0 + k1);
            b1N = load_b_global(Bb1 + k1);
            b2N = load_b_global(Bb2 + k1);
            b3N = load_b_global(Bb3 + k1);
        }
        acc[0] = __builtin_amdgcn_wmma_f32_16x16x32_bf16(false, aC, false, b0C.v, (short)0, acc[0], false, false);
        acc[1] = __builtin_amdgcn_wmma_f32_16x16x32_bf16(false, aC, false, b1C.v, (short)0, acc[1], false, false);
        acc[2] = __builtin_amdgcn_wmma_f32_16x16x32_bf16(false, aC, false, b2C.v, (short)0, acc[2], false, false);
        acc[3] = __builtin_amdgcn_wmma_f32_16x16x32_bf16(false, aC, false, b3C.v, (short)0, acc[3], false, false);
        aC = aN; b0C = b0N; b1C = b1N; b2C = b2N; b3C = b3N;
    }

#pragma unroll
    for (int j = 0; j < 4; ++j) {
        const int col = (nstrip * 4 + j) * 16 + nl;
        const float bsum = bias[col];
#pragma unroll
        for (int r = 0; r < 8; ++r) {
            const size_t row = (size_t)mstrip * 16 + half * 8 + r;
            XP[row * HID + col] = acc[j][r] + bsum;
        }
    }
}

// ---------------------------------------------------------------------------
// Phase 2: persistent scan kernel. 64 blocks x 256 threads (8 waves).
// Block = one 16-column strip (nt); wave = one 16-row tile (mt).
// WhT strip (16 x 1024 bf16 = 32KB) pulled into LDS with the CDNA5 async
// global->LDS engine (ASYNCcnt), padded rows to dodge bank conflicts.
// Per step: h_new = tanh(XP_t + h @ Wh), grid barrier via global atomic.
// Double-buffered fragments + two alternating accumulators.
// ---------------------------------------------------------------------------
__global__ __launch_bounds__(256) void rnn_scan_kernel(
    const float* __restrict__ XP, const unsigned short* __restrict__ WhT,
    unsigned short* __restrict__ hb0, unsigned short* __restrict__ hb1,
    float* __restrict__ out, unsigned int* __restrict__ counter)
{
    __shared__ uint4 ldsWh[16 * 129];          // 16 rows x (128 + 1 pad) uint4 = 33 KB

    const int nt   = blockIdx.x;               // 0..63
    const int wv   = threadIdx.x >> 5;         // 0..7  -> M tile
    const int lane = threadIdx.x & 31;
    const int half = lane >> 4;
    const int nl   = lane & 15;

    // Async-copy WhT rows [nt*16, nt*16+16) -- contiguous 32KB -- into padded LDS.
    // GLOBAL_LOAD_ASYNC_TO_LDS_B128: vdst = LDS byte offset, vaddr = 64-bit global
    // address; tracked with ASYNCcnt, drained with s_wait_asynccnt.
    {
        const uint4* src = (const uint4*)(WhT + (size_t)nt * 16 * HID);
        for (int i = threadIdx.x; i < 16 * 128; i += 256) {
            const unsigned ldsoff = (unsigned)(size_t)(&ldsWh[(i >> 7) * 129 + (i & 127)]);
            const uint4* gaddr = src + i;
            asm volatile("global_load_async_to_lds_b128 %0, %1, off"
                         :: "v"(ldsoff), "v"(gaddr) : "memory");
        }
        asm volatile("s_wait_asynccnt 0x0" ::: "memory");
    }
    __syncthreads();

    const int ldsbase = nl * 129;              // lane's B column row in LDS

    for (int t = 0; t < T_STEPS; ++t) {
        const unsigned short* hcur = (t & 1) ? hb1 : hb0;
        unsigned short*       hnxt = (t & 1) ? hb0 : hb1;

        const unsigned short* Arow = hcur + (size_t)(wv * 16 + nl) * HID;
        const float* xpt = XP + ((size_t)t * BATCH + wv * 16 + half * 8) * HID + nt * 16 + nl;
        if (t < T_STEPS - 1)
            __builtin_prefetch(xpt + (size_t)BATCH * HID, 0, 1);  // next step's XP tile

        v8f acc0 = {}, acc1 = {};

        // prologue: chunk 0
        Frag16 aC, bC;
        aC.q[0] = *(const uint4*)(Arow + half * 8);
        aC.q[1] = *(const uint4*)(Arow + 16 + half * 8);
        bC.q[0] = ldsWh[ldsbase + half];
        bC.q[1] = ldsWh[ldsbase + half + 2];

#pragma unroll
        for (int c = 0; c < 32; ++c) {         // K = 1024 in chunks of 32
            Frag16 aN = {}, bN = {};
            if (c < 31) {
                const int k1 = (c + 1) * 32;
                aN.q[0] = *(const uint4*)(Arow + k1 + half * 8);
                aN.q[1] = *(const uint4*)(Arow + k1 + 16 + half * 8);
                bN.q[0] = ldsWh[ldsbase + (c + 1) * 4 + half];
                bN.q[1] = ldsWh[ldsbase + (c + 1) * 4 + half + 2];
            }
            if (c & 1)
                acc1 = __builtin_amdgcn_wmma_f32_16x16x32_bf16(false, aC.v, false, bC.v, (short)0, acc1, false, false);
            else
                acc0 = __builtin_amdgcn_wmma_f32_16x16x32_bf16(false, aC.v, false, bC.v, (short)0, acc0, false, false);
            aC = aN; bC = bN;
        }
        const v8f acc = acc0 + acc1;

#pragma unroll
        for (int r = 0; r < 8; ++r) {
            const float v = tanhf(acc[r] + xpt[(size_t)r * HID]);
            const size_t oidx = (size_t)(wv * 16 + half * 8 + r) * HID + nt * 16 + nl;
            reinterpret_cast<__hip_bfloat16*>(hnxt)[oidx] = __float2bfloat16(v);
            if (t == T_STEPS - 1) out[oidx] = v;
        }

        // ---- grid-wide barrier (all 64 blocks) ----
        __threadfence();
        __syncthreads();
        if (threadIdx.x == 0) {
            __hip_atomic_fetch_add(counter, 1u, __ATOMIC_RELEASE, __HIP_MEMORY_SCOPE_AGENT);
            const unsigned int target = (unsigned int)(t + 1) * 64u;
            while (__hip_atomic_load(counter, __ATOMIC_ACQUIRE, __HIP_MEMORY_SCOPE_AGENT) < target)
                __builtin_amdgcn_s_sleep(2);
        }
        __syncthreads();
        __threadfence();
    }
}

// ---------------------------------------------------------------------------
// Host launcher
// ---------------------------------------------------------------------------
extern "C" void kernel_launch(void* const* d_in, const int* in_sizes, int n_in,
                              void* d_out, int out_size, void* d_ws, size_t ws_size,
                              hipStream_t stream)
{
    const float* x  = (const float*)d_in[0];   // [512,128,512]
    const float* h0 = (const float*)d_in[1];   // [128,1024]
    const float* Wx = (const float*)d_in[2];   // [512,1024]
    const float* bx = (const float*)d_in[3];   // [1024]
    const float* Wh = (const float*)d_in[4];   // [1024,1024]
    const float* bh = (const float*)d_in[5];   // [1024]

    char* ws = (char*)d_ws;
    // workspace layout (all offsets 256B aligned)
    unsigned int*   counter = (unsigned int*)(ws + 0);        //   256 B
    float*          bias    = (float*)(ws + 256);             //   4 KB
    unsigned short* hb0     = (unsigned short*)(ws + 4352);   // 256 KB bf16 h ping
    unsigned short* hb1     = (unsigned short*)(ws + 266496); // 256 KB bf16 h pong
    unsigned short* WxT     = (unsigned short*)(ws + 528640); //   1 MB bf16 Wx^T [1024][512]
    unsigned short* WhT     = (unsigned short*)(ws + 1577216);//   2 MB bf16 Wh^T [1024][1024]
    float*          XP      = (float*)(ws + 3674368);         // 256 MB f32 XP [65536][1024]

    hipMemsetAsync(counter, 0, 256, stream);   // reset grid barrier (capture-safe)

    rnn_prep_kernel<<<4096, 256, 0, stream>>>(Wx, bx, Wh, bh, h0, bias, hb0, WxT, WhT);
    rnn_gemm1_kernel<<<8192, 256, 0, stream>>>(x, WxT, bias, XP);
    rnn_scan_kernel<<<64, 256, 0, stream>>>(XP, WhT, hb0, hb1, (float*)d_out, counter);
}
